// PerformerAttention_2027224564001
// MI455X (gfx1250) — matrix operand order repaired and checked by reading, not compiled, not run on hardware
//
#include <hip/hip_runtime.h>
#include <hip/hip_bf16.h>

// CDNA5 / gfx1250 Performer (linear) attention.
// B=4,H=16,S=4096,D=E=64. Two-phase: kv = K'^T V reduction, then num = Q' kv.
// bf16 WMMA (v_wmma_f32_16x16x32_bf16) with fp32 accumulation for both GEMMs.

typedef __attribute__((ext_vector_type(16))) __bf16 v16bf;
typedef __attribute__((ext_vector_type(8)))  float  v8f;

#define S_LEN 4096
#define NHEAD 64          // B*H
#define HPERB 16
#define SPLIT 4
#define CHUNK (S_LEN / SPLIT)

// A-matrix (16x32, 16-bit) K-index for fragment slot i, lane-half h  (ISA 7.12.2)
__device__ __forceinline__ int kA(int i, int h) { return (i < 8) ? (i + 8 * h) : (i + 8 + 8 * h); }
// B-matrix (32x16, 16-bit) K-index: same order as A (slots 0-7 -> K=8h+i, slots 8-15 -> K=16+8h+(i-8))
__device__ __forceinline__ int kB(int i, int h) { return (i < 8) ? (i + 8 * h) : (i + 8 + 8 * h); }

// ---------------------------------------------------------------------------
// Phase 1: per (head, S-split): partial kv[64][64] = (relu(K)+eps)^T * V and
// partial ksum[64]. 128 threads = 4 waves; wave w owns d-rows [16w,16w+16).
// ---------------------------------------------------------------------------
__global__ __launch_bounds__(128) void performer_phase1(
    const float* __restrict__ Kp, const float* __restrict__ Vp,
    const float* __restrict__ Mp, float* __restrict__ kvPart,
    float* __restrict__ ksPart)
{
  __shared__ float Kl[32][68];   // padded to dodge bank conflicts
  __shared__ float Vl[32][68];
  __shared__ float red[128];

  const int t     = threadIdx.x;
  const int head  = blockIdx.x >> 2;          // /SPLIT
  const int split = blockIdx.x & (SPLIT - 1);
  const int bIdx  = head / HPERB;
  const int lane  = t & 31;
  const int wave  = t >> 5;
  const int m     = lane & 15;
  const int hh    = lane >> 4;
  const int dBase = wave * 16;
  const size_t base = (size_t)head * S_LEN * 64;
  const int sBeg  = split * CHUNK;

  v8f acc[4] = {};
  float ksLocal = 0.f;

  for (int s0 = sBeg; s0 < sBeg + CHUNK; s0 += 32) {
    // Cooperative coalesced load of K,V tile [32 s x 64 d]; thread t always
    // touches d = t%64, so ksum accumulates per-thread for a fixed column.
    #pragma unroll
    for (int it = 0; it < 16; ++it) {
      int idx = t + 128 * it;
      int sr  = idx >> 6;
      int d   = idx & 63;
      int sg  = s0 + sr;
      float kraw = Kp[base + (size_t)sg * 64 + d];
      float kval = (kraw > 0.f ? kraw : 0.f) + 0.001f;
      kval *= Mp[bIdx * S_LEN + sg];
      Kl[sr][d] = kval;
      ksLocal  += kval;
      Vl[sr][d] = Vp[base + (size_t)sg * 64 + d];
    }
    if (s0 + 32 < sBeg + CHUNK) {
      __builtin_prefetch(&Kp[base + (size_t)(s0 + 32) * 64 + t], 0, 1);
      __builtin_prefetch(&Vp[base + (size_t)(s0 + 32) * 64 + t], 0, 1);
    }
    __syncthreads();

    // A = K^T tile: element (M = d-within-wave, K = s-within-tile)
    v16bf a;
    #pragma unroll
    for (int i = 0; i < 16; ++i) a[i] = (__bf16)Kl[kA(i, hh)][dBase + m];

    #pragma unroll
    for (int n = 0; n < 4; ++n) {
      // B = V tile: element (K = s-within-tile, N = e-within-n-tile)
      v16bf b;
      #pragma unroll
      for (int i = 0; i < 16; ++i) b[i] = (__bf16)Vl[kB(i, hh)][n * 16 + m];
      acc[n] = __builtin_amdgcn_wmma_f32_16x16x32_bf16(
          false, a, false, b, (short)0, acc[n], false, false);
    }
    __syncthreads();
  }

  // C/D layout: VGPR r -> M = r + 8*(lane/16), N = lane%16
  const size_t kvOff = (size_t)(split * NHEAD + head) * 4096;
  #pragma unroll
  for (int n = 0; n < 4; ++n)
    #pragma unroll
    for (int r = 0; r < 8; ++r)
      kvPart[kvOff + (size_t)(dBase + r + 8 * hh) * 64 + n * 16 + m] = acc[n][r];

  red[t] = ksLocal;
  __syncthreads();
  if (t < 64) ksPart[(split * NHEAD + head) * 64 + t] = red[t] + red[t + 64];
}

// ---------------------------------------------------------------------------
// Phase 2: per (head, 64-row S tile): num = (relu(Q)+eps) * kv via WMMA,
// den = q . ksum (fp32), out = num/den.
// ---------------------------------------------------------------------------
__global__ __launch_bounds__(128) void performer_phase2(
    const float* __restrict__ Qp, const float* __restrict__ kvPart,
    const float* __restrict__ ksPart, float* __restrict__ out)
{
  __shared__ float qs[64][68];
  __shared__ float kvs[64][68];
  __shared__ float ks[64];
  __shared__ float den[64];

  const int t       = threadIdx.x;
  const int head    = blockIdx.x >> 6;
  const int tile    = blockIdx.x & 63;
  const int rowBase = tile * 64;
  const size_t qbase = (size_t)head * S_LEN * 64 + (size_t)rowBase * 64;

  #pragma unroll
  for (int it = 0; it < 32; ++it) {
    int idx = t + 128 * it;
    float q = Qp[qbase + idx];
    qs[idx >> 6][idx & 63] = (q > 0.f ? q : 0.f) + 0.001f;
  }
  #pragma unroll
  for (int it = 0; it < 32; ++it) {
    int idx = t + 128 * it;
    float v = 0.f;
    #pragma unroll
    for (int sp = 0; sp < SPLIT; ++sp)
      v += kvPart[(size_t)(sp * NHEAD + head) * 4096 + idx];
    kvs[idx >> 6][idx & 63] = v;
  }
  if (t < 64) {
    float v = 0.f;
    #pragma unroll
    for (int sp = 0; sp < SPLIT; ++sp) v += ksPart[(sp * NHEAD + head) * 64 + t];
    ks[t] = v;
  }
  __syncthreads();
  if (t < 64) {
    float s = 0.f;
    #pragma unroll
    for (int d = 0; d < 64; ++d) s += qs[t][d] * ks[d];
    den[t] = s;
  }
  __syncthreads();

  const int lane = t & 31, wave = t >> 5;
  const int m = lane & 15, hh = lane >> 4;
  const int mRow = wave * 16;

  // A = Q tile (16 rows x 64 d) as two 16x32 pieces
  v16bf a0, a1;
  #pragma unroll
  for (int i = 0; i < 16; ++i) {
    a0[i] = (__bf16)qs[mRow + m][kA(i, hh)];
    a1[i] = (__bf16)qs[mRow + m][kA(i, hh) + 32];
  }

  v8f acc[4];
  #pragma unroll
  for (int n = 0; n < 4; ++n) {
    v16bf b0, b1;
    #pragma unroll
    for (int i = 0; i < 16; ++i) {
      b0[i] = (__bf16)kvs[kB(i, hh)][n * 16 + m];
      b1[i] = (__bf16)kvs[kB(i, hh) + 32][n * 16 + m];
    }
    v8f c = {};
    c = __builtin_amdgcn_wmma_f32_16x16x32_bf16(
        false, a0, false, b0, (short)0, c, false, false);
    acc[n] = __builtin_amdgcn_wmma_f32_16x16x32_bf16(
        false, a1, false, b1, (short)0, c, false, false);
  }

  #pragma unroll
  for (int n = 0; n < 4; ++n)
    #pragma unroll
    for (int r = 0; r < 8; ++r) {
      int sl = mRow + r + 8 * hh;
      out[((size_t)head * S_LEN + rowBase + sl) * 64 + n * 16 + m] =
          acc[n][r] / den[sl];
    }
}

// ---------------------------------------------------------------------------
extern "C" void kernel_launch(void* const* d_in, const int* in_sizes, int n_in,
                              void* d_out, int out_size, void* d_ws, size_t ws_size,
                              hipStream_t stream) {
  const float* Q  = (const float*)d_in[0];
  const float* K  = (const float*)d_in[1];
  const float* V  = (const float*)d_in[2];
  const float* Mk = (const float*)d_in[3];

  float* ws     = (float*)d_ws;
  float* kvPart = ws;                                   // SPLIT*NHEAD*64*64 floats
  float* ksPart = ws + (size_t)SPLIT * NHEAD * 4096;    // SPLIT*NHEAD*64 floats

  performer_phase1<<<NHEAD * SPLIT, 128, 0, stream>>>(K, V, Mk, kvPart, ksPart);
  performer_phase2<<<NHEAD * (S_LEN / 64), 128, 0, stream>>>(Q, kvPart, ksPart,
                                                            (float*)d_out);
}
